// TransformerCRF_58841051955735
// MI455X (gfx1250) — compile-verified, matrix operand order checked by reading, not run on hardware
//
#include <hip/hip_runtime.h>
#include <hip/hip_bf16.h>
#include <math.h>

// Problem constants (match reference)
#define B_  32
#define S_  512
#define D_  512
#define H_  8
#define DH_ 64
#define L_  2
#define FF_ 2048
#define T_  32

typedef __attribute__((ext_vector_type(16))) __bf16 v16bf;
typedef __attribute__((ext_vector_type(8)))  float  v8f;
typedef __attribute__((ext_vector_type(4)))  unsigned int tdm_v4u;
typedef __attribute__((ext_vector_type(8)))  int          tdm_v8i;
typedef __attribute__((ext_vector_type(4)))  int          tdm_v4i;

#define WMMA_BF16(a,b,cacc) \
  __builtin_amdgcn_wmma_f32_16x16x32_bf16(false,(a),false,(b),(short)0,(cacc),false,false)

__device__ inline unsigned short f2bf(float f) {
  // round-to-nearest-even f32 -> bf16
  unsigned int u = __float_as_uint(f);
  u += 0x7FFFu + ((u >> 16) & 1u);
  return (unsigned short)(u >> 16);
}

// 16-bit WMMA per-lane fragment: lanes 0-15 hold K=0..7,16..23; lanes 16-31
// hold K=8..15,24..31 -> two contiguous 16-byte runs => two b128 loads.
union FragV { uint4 q[2]; v16bf v; };

// Fragment from bf16 tile (LDS or global), row-major, row length `ld` elems.
__device__ inline v16bf frag16(const unsigned short* base, int row0, int ld, int lane) {
  const int kh = (lane >> 4) << 3;
  const unsigned short* p = base + (size_t)(row0 + (lane & 15)) * ld;
  FragV f;
  f.q[0] = *(const uint4*)(p + kh);
  f.q[1] = *(const uint4*)(p + 16 + kh);
  return f.v;
}

// Fragment from a per-lane base pointer (lane-fixed row already folded in).
__device__ inline v16bf frag16p(const unsigned short* p, int lane) {
  const int kh = (lane >> 4) << 3;
  FragV f;
  f.q[0] = *(const uint4*)(p + kh);
  f.q[1] = *(const uint4*)(p + 16 + kh);
  return f.v;
}

// ---------------------------------------------------------------------------
// Tensor Data Mover: async 2-D bf16 tile load global -> LDS (TENSORcnt).
// ---------------------------------------------------------------------------
__device__ inline void tdm_load_2d(unsigned lds_off, const void* gaddr,
                                   unsigned tensor_d0, unsigned tensor_d1,
                                   unsigned long long d0_stride,
                                   unsigned tile_d0, unsigned tile_d1)
{
  unsigned long long ga = (unsigned long long)(uintptr_t)gaddr;
  tdm_v4u g0;
  g0[0] = 1u;                                        // count=1 (user descriptor)
  g0[1] = lds_off;                                   // lds_addr (bytes)
  g0[2] = (unsigned)(ga & 0xFFFFFFFFu);              // global_addr[31:0]
  g0[3] = (unsigned)((ga >> 32) & 0x01FFFFFFu) | (2u << 30);  // addr[56:32]|type=2
  tdm_v8i g1;
  g1[0] = (int)(1u << 16);                           // data_size=1 (2 bytes)
  g1[1] = (int)((tensor_d0 & 0xFFFFu) << 16);        // tensor_dim0[15:0] @63:48
  g1[2] = (int)((tensor_d0 >> 16) | ((tensor_d1 & 0xFFFFu) << 16));
  g1[3] = (int)((tensor_d1 >> 16) | (tile_d0 << 16));// tile_dim0 @127:112
  g1[4] = (int)(tile_d1 & 0xFFFFu);                  // tile_dim1; tile_dim2=0
  g1[5] = (int)(unsigned)(d0_stride & 0xFFFFFFFFu);  // dim0_stride[31:0]
  g1[6] = (int)(unsigned)((d0_stride >> 32) & 0xFFFFu);
  g1[7] = 0;
  tdm_v4i z = {0, 0, 0, 0};
#if defined(__clang_major__) && (__clang_major__ >= 23)
  tdm_v8i z8 = {0, 0, 0, 0, 0, 0, 0, 0};
  __builtin_amdgcn_tensor_load_to_lds(g0, g1, z, z, z8, 0);
#else
  __builtin_amdgcn_tensor_load_to_lds(g0, g1, z, z, 0);
#endif
}

// ---------------------------------------------------------------------------
// f32 -> bf16 elementwise convert (for weights). n is a multiple of 4.
// ---------------------------------------------------------------------------
__global__ __launch_bounds__(256) void cvt_bf16_kernel(
    const float* __restrict__ in, unsigned short* __restrict__ out, int n)
{
  int i = (blockIdx.x * 256 + threadIdx.x) * 4;
  if (i + 3 < n) {
    float4 v = *(const float4*)(in + i);
    uint2 w;
    w.x = (unsigned)f2bf(v.x) | ((unsigned)f2bf(v.y) << 16);
    w.y = (unsigned)f2bf(v.z) | ((unsigned)f2bf(v.w) << 16);
    *(uint2*)(out + i) = w;
  }
}

// ---------------------------------------------------------------------------
// C[M,N] = A[M,K] @ W[N,K]^T + bias[N]; A,W bf16; optional ReLU; writes f32
// C and/or bf16 C (either may be null).
// 128x64 block tile, 4 waves; wave = 32 rows x 64 cols (2x4 WMMA accums).
// K staged 32-deep via double-buffered TDM: loads for tile k+1 are issued
// before computing tile k; s_wait_tensorcnt<=2 retires only the current
// buffer's two loads (TENSORcnt completes in-order), overlapping DMA+WMMA.
// ---------------------------------------------------------------------------
__global__ __launch_bounds__(128) void gemm_bt_kernel(
    const unsigned short* __restrict__ A, const unsigned short* __restrict__ W,
    const float* __restrict__ bias, float* __restrict__ Cf,
    unsigned short* __restrict__ Cbf, int M, int N, int K, int relu)
{
  __shared__ unsigned short As[2][128 * 32];  // 2 x 8KB
  __shared__ unsigned short Ws[2][64 * 32];   // 2 x 4KB
  const int tid = threadIdx.x, lane = tid & 31, wave = tid >> 5;
  const int m0 = blockIdx.y * 128, n0 = blockIdx.x * 64;
  const int wm = wave * 32;
  v8f acc[2][4] = {};

  if (wave == 0) {
    tdm_load_2d((unsigned)(uintptr_t)&As[0][0], A + (size_t)m0 * K,
                32u, (unsigned)(M - m0), (unsigned)K, 32u, 128u);
    tdm_load_2d((unsigned)(uintptr_t)&Ws[0][0], W + (size_t)n0 * K,
                32u, (unsigned)(N - n0), (unsigned)K, 32u, 64u);
  }

  int buf = 0;
  for (int k0 = 0; k0 < K; k0 += 32) {
    if (wave == 0) {
      if (k0 + 32 < K) {
        tdm_load_2d((unsigned)(uintptr_t)&As[buf ^ 1][0],
                    A + (size_t)m0 * K + k0 + 32,
                    32u, (unsigned)(M - m0), (unsigned)K, 32u, 128u);
        tdm_load_2d((unsigned)(uintptr_t)&Ws[buf ^ 1][0],
                    W + (size_t)n0 * K + k0 + 32,
                    32u, (unsigned)(N - n0), (unsigned)K, 32u, 64u);
        __builtin_amdgcn_s_wait_tensorcnt(2);  // current buffer retired
      } else {
        __builtin_amdgcn_s_wait_tensorcnt(0);
      }
    }
    __syncthreads();

    const unsigned short* as = &As[buf][0];
    const unsigned short* ws = &Ws[buf][0];
    v16bf aF[2], bF[4];
    aF[0] = frag16(as, wm,      32, lane);
    aF[1] = frag16(as, wm + 16, 32, lane);
#pragma unroll
    for (int j = 0; j < 4; ++j) bF[j] = frag16(ws, j * 16, 32, lane);
#pragma unroll
    for (int i = 0; i < 2; ++i)
#pragma unroll
      for (int j = 0; j < 4; ++j)
        acc[i][j] = WMMA_BF16(aF[i], bF[j], acc[i][j]);
    __syncthreads();  // all waves done with `buf` before TDM refills it
    buf ^= 1;
  }

  // C layout: VGPR r, lane-group g -> row 8g+r; lane&15 -> col.
  const int g = lane >> 4, c = lane & 15;
#pragma unroll
  for (int i = 0; i < 2; ++i)
#pragma unroll
    for (int j = 0; j < 4; ++j)
#pragma unroll
      for (int r = 0; r < 8; ++r) {
        int m = m0 + wm + i * 16 + 8 * g + r;
        int n = n0 + j * 16 + c;
        if (m < M && n < N) {
          float v = acc[i][j][r] + bias[n];
          if (relu) v = fmaxf(v, 0.f);
          if (Cf)  Cf[(size_t)m * N + n] = v;
          if (Cbf) Cbf[(size_t)m * N + n] = f2bf(v);
        }
      }
}

// ---------------------------------------------------------------------------
// Flash attention on the bf16 qkv copy: grid (S/64, B*H), 4 waves/block, one
// wave = 16 query rows. K tile staged via TDM (overlapped with the manual
// transposed V staging); tiles shared by all 4 waves. mask is all-ones.
// Output written as bf16 (only consumed by the Wo GEMM).
// ---------------------------------------------------------------------------
__global__ __launch_bounds__(128) void attn_kernel(
    const unsigned short* __restrict__ qkvbf, unsigned short* __restrict__ obf)
{
  __shared__ unsigned short Kt[32 * 64];      // [kv][d] row-major (TDM)
  __shared__ unsigned short Vt[64 * 32];      // [d][kv] (transposed, manual)
  __shared__ unsigned short Pl[4][16 * 32];   // per-wave P tiles
  const int tid = threadIdx.x, lane = tid & 31, wave = tid >> 5;
  const int bh = blockIdx.y, b = bh >> 3, h = bh & 7;
  const int q0 = blockIdx.x * 64 + wave * 16;
  const int c = lane & 15, g = lane >> 4;
  const int RS = 3 * D_;
  const unsigned short* base = qkvbf + (size_t)b * S_ * RS;
  const unsigned short* Qb = base + h * DH_;
  const unsigned short* Kb = base + D_ + h * DH_;
  const unsigned short* Vb = base + 2 * D_ + h * DH_;

  // Q fragments: two b128 loads each, straight from global bf16.
  const unsigned short* pQ = Qb + (size_t)(q0 + c) * RS;
  v16bf aQ0 = frag16p(pQ, lane);
  v16bf aQ1 = frag16p(pQ + 32, lane);

  v8f Oacc[4] = {};
  float rowm[8], rowl[8];
#pragma unroll
  for (int r = 0; r < 8; ++r) { rowm[r] = -1e30f; rowl[r] = 0.f; }

  // Staging map: thread covers 16 consecutive d within one kv row.
  const int skv = tid >> 2, sd = (tid & 3) * 16;

  for (int kv0 = 0; kv0 < S_; kv0 += 32) {
    if (wave == 0)  // K tile: 32 rows x 64 cols, row stride RS
      tdm_load_2d((unsigned)(uintptr_t)&Kt[0], Kb + (size_t)kv0 * RS,
                  64u, (unsigned)(S_ - kv0), (unsigned)RS, 64u, 32u);
    {
      // V tile transposed (TDM cannot transpose): overlaps with the DMA.
      const unsigned short* vs = Vb + (size_t)(kv0 + skv) * RS + sd;
#pragma unroll
      for (int e = 0; e < 16; ++e) Vt[(sd + e) * 32 + skv] = vs[e];
    }
    if (wave == 0) __builtin_amdgcn_s_wait_tensorcnt(0);
    __syncthreads();

    // scores = Q K^T / 8 : DH=64 -> two K-steps per C tile.
    v8f sc0 = {}, sc1 = {};
    sc0 = WMMA_BF16(aQ0, frag16(Kt,      0,  64, lane), sc0);
    sc0 = WMMA_BF16(aQ1, frag16(Kt + 32, 0,  64, lane), sc0);
    sc1 = WMMA_BF16(aQ0, frag16(Kt,      16, 64, lane), sc1);
    sc1 = WMMA_BF16(aQ1, frag16(Kt + 32, 16, 64, lane), sc1);
    sc0 = sc0 * 0.125f;  // 1/sqrt(DH)
    sc1 = sc1 * 0.125f;

    // Online softmax (row reductions across the 16 column lanes per row).
    float nm[8], al[8];
#pragma unroll
    for (int r = 0; r < 8; ++r) {
      float v = fmaxf(sc0[r], sc1[r]);
#pragma unroll
      for (int off = 8; off; off >>= 1) v = fmaxf(v, __shfl_xor(v, off, 16));
      nm[r] = fmaxf(rowm[r], v);
      al[r] = __expf(rowm[r] - nm[r]);
      rowm[r] = nm[r];
    }
#pragma unroll
    for (int r = 0; r < 8; ++r) {
      sc0[r] = __expf(sc0[r] - nm[r]);
      sc1[r] = __expf(sc1[r] - nm[r]);
      float t = sc0[r] + sc1[r];
#pragma unroll
      for (int off = 8; off; off >>= 1) t += __shfl_xor(t, off, 16);
      rowl[r] = rowl[r] * al[r] + t;
    }
#pragma unroll
    for (int t = 0; t < 4; ++t)
#pragma unroll
      for (int r = 0; r < 8; ++r) Oacc[t][r] *= al[r];

    // P (C layout) -> LDS bf16 -> reload in A layout (two b128 loads).
    unsigned short* pl = &Pl[wave][0];
#pragma unroll
    for (int r = 0; r < 8; ++r) {
      pl[(8 * g + r) * 32 + c]      = f2bf(sc0[r]);
      pl[(8 * g + r) * 32 + 16 + c] = f2bf(sc1[r]);
    }
    __syncthreads();
    v16bf aP = frag16(pl, 0, 32, lane);
#pragma unroll
    for (int t = 0; t < 4; ++t) {
      // V B-fragment from transposed tile: row = d column, contraction = kv.
      v16bf bV = frag16(Vt, t * 16, 32, lane);
      Oacc[t] = WMMA_BF16(aP, bV, Oacc[t]);
    }
    __syncthreads();  // protect Kt/Vt/Pl before next stage
  }

#pragma unroll
  for (int t = 0; t < 4; ++t)
#pragma unroll
    for (int r = 0; r < 8; ++r) {
      int q = q0 + 8 * g + r;
      obf[((size_t)b * S_ + q) * D_ + h * DH_ + t * 16 + c] =
          f2bf(Oacc[t][r] / rowl[r]);
    }
}

// ---------------------------------------------------------------------------
// out = LayerNorm(x + y) * scale + bias; writes f32 and bf16 copies.
// ---------------------------------------------------------------------------
__global__ __launch_bounds__(256) void add_ln_kernel(
    const float* __restrict__ x, const float* __restrict__ y,
    const float* __restrict__ sc, const float* __restrict__ bi,
    float* __restrict__ out, unsigned short* __restrict__ outbf)
{
  __shared__ float red[256];
  const int row = blockIdx.x, tid = threadIdx.x;
  const size_t off = (size_t)row * D_;
  float v0 = x[off + tid]       + y[off + tid];
  float v1 = x[off + tid + 256] + y[off + tid + 256];
  red[tid] = v0 + v1;
  __syncthreads();
  for (int st = 128; st; st >>= 1) { if (tid < st) red[tid] += red[tid + st]; __syncthreads(); }
  float mu = red[0] / (float)D_;
  __syncthreads();
  float d0 = v0 - mu, d1 = v1 - mu;
  red[tid] = d0 * d0 + d1 * d1;
  __syncthreads();
  for (int st = 128; st; st >>= 1) { if (tid < st) red[tid] += red[tid + st]; __syncthreads(); }
  float rinv = rsqrtf(red[0] / (float)D_ + 1e-5f);
  float r0 = d0 * rinv * sc[tid]       + bi[tid];
  float r1 = d1 * rinv * sc[tid + 256] + bi[tid + 256];
  out[off + tid]         = r0;
  out[off + tid + 256]   = r1;
  outbf[off + tid]       = f2bf(r0);
  outbf[off + tid + 256] = f2bf(r1);
}

// ---------------------------------------------------------------------------
// x[b,s,:] = emb[token] + sinusoidal PE(s,:)  (f32 + bf16 copies)
// ---------------------------------------------------------------------------
__global__ __launch_bounds__(256) void embed_kernel(
    const int* __restrict__ tokens, const float* __restrict__ emb,
    float* __restrict__ x, unsigned short* __restrict__ xbf)
{
  const int bs = blockIdx.x, s = bs & (S_ - 1);
  const int tok = tokens[bs];
  for (int d = threadIdx.x; d < D_; d += 256) {
    int i = d >> 1;
    float div = __expf((float)(2 * i) * (-9.210340371976184f / (float)D_));
    float ang = (float)s * div;
    float pe = (d & 1) ? __cosf(ang) : __sinf(ang);
    float v = emb[(size_t)tok * D_ + d] + pe;
    x[(size_t)bs * D_ + d] = v;
    xbf[(size_t)bs * D_ + d] = f2bf(v);
  }
}

// ---------------------------------------------------------------------------
// CRF gold-path score (mask all-true -> last position = S-1).
// ---------------------------------------------------------------------------
__global__ __launch_bounds__(32) void crf_score_kernel(
    const float* __restrict__ emis, const int* __restrict__ tags,
    const float* __restrict__ trans, const float* __restrict__ start_t,
    const float* __restrict__ end_t, float* __restrict__ score)
{
  const int b = threadIdx.x;
  int prev = tags[(size_t)b * S_];
  float sc = start_t[prev] + emis[((size_t)b * S_) * T_ + prev];
  for (int s = 1; s < S_; ++s) {
    int tg = tags[(size_t)b * S_ + s];
    sc += trans[prev * T_ + tg] + emis[((size_t)b * S_ + s) * T_ + tg];
    prev = tg;
  }
  score[b] = sc + end_t[prev];
}

// ---------------------------------------------------------------------------
// CRF forward algorithm: one wave per batch row, lane j = alpha[j] (T=32 ==
// wave32). trans cached in LDS; alpha broadcast via __shfl.
// ---------------------------------------------------------------------------
__global__ __launch_bounds__(32) void crf_fwd_kernel(
    const float* __restrict__ emis, const float* __restrict__ trans,
    const float* __restrict__ start_t, const float* __restrict__ end_t,
    float* __restrict__ logz)
{
  __shared__ float tr[T_ * T_];
  const int j = threadIdx.x, b = blockIdx.x;
  for (int i = 0; i < T_; ++i) tr[i * T_ + j] = trans[i * T_ + j];
  __syncthreads();

  float alpha = start_t[j] + emis[((size_t)b * S_) * T_ + j];
  for (int s = 1; s < S_; ++s) {
    float m = -1e30f;
    for (int i = 0; i < T_; ++i) {
      float ai = __shfl(alpha, i, 32);
      m = fmaxf(m, ai + tr[i * T_ + j]);
    }
    float ssum = 0.f;
    for (int i = 0; i < T_; ++i) {
      float ai = __shfl(alpha, i, 32);
      ssum += __expf(ai + tr[i * T_ + j] - m);
    }
    alpha = m + __logf(ssum) + emis[((size_t)b * S_ + s) * T_ + j];
  }
  float v = alpha + end_t[j];
  float m = v;
#pragma unroll
  for (int off = 16; off; off >>= 1) m = fmaxf(m, __shfl_xor(m, off, 32));
  float e = __expf(v - m);
#pragma unroll
  for (int off = 16; off; off >>= 1) e += __shfl_xor(e, off, 32);
  if (j == 0) logz[b] = m + __logf(e);
}

__global__ __launch_bounds__(32) void crf_loss_kernel(
    const float* __restrict__ logz, const float* __restrict__ score,
    float* __restrict__ out)
{
  float v = logz[threadIdx.x] - score[threadIdx.x];
#pragma unroll
  for (int off = 16; off; off >>= 1) v += __shfl_xor(v, off, 32);
  if (threadIdx.x == 0) out[0] = v / (float)B_;
}

// ---------------------------------------------------------------------------
// Host orchestration. Inputs (setup_inputs order):
// 0 tokens 1 tags 2 mask(all-true; unused) 3 emb 4 Wqkv 5 bqkv 6 Wo 7 bo
// 8 ln1_s 9 ln1_b 10 W1 11 b1 12 W2 13 b2 14 ln2_s 15 ln2_b 16 Wtag 17 btag
// 18 trans 19 start_t 20 end_t.  Output: scalar f32 loss.
// ---------------------------------------------------------------------------
extern "C" void kernel_launch(void* const* d_in, const int* in_sizes, int n_in,
                              void* d_out, int out_size, void* d_ws, size_t ws_size,
                              hipStream_t stream) {
  (void)in_sizes; (void)n_in; (void)out_size; (void)ws_size;
  const int*   tokens = (const int*)d_in[0];
  const int*   tags   = (const int*)d_in[1];
  const float* emb    = (const float*)d_in[3];
  const float* Wqkv   = (const float*)d_in[4];
  const float* bqkv   = (const float*)d_in[5];
  const float* Wo     = (const float*)d_in[6];
  const float* bo     = (const float*)d_in[7];
  const float* ln1_s  = (const float*)d_in[8];
  const float* ln1_b  = (const float*)d_in[9];
  const float* W1     = (const float*)d_in[10];
  const float* b1     = (const float*)d_in[11];
  const float* W2     = (const float*)d_in[12];
  const float* b2     = (const float*)d_in[13];
  const float* ln2_s  = (const float*)d_in[14];
  const float* ln2_b  = (const float*)d_in[15];
  const float* Wtag   = (const float*)d_in[16];
  const float* btag   = (const float*)d_in[17];
  const float* trans  = (const float*)d_in[18];
  const float* start_t= (const float*)d_in[19];
  const float* end_t  = (const float*)d_in[20];

  // Workspace carving (256B aligned slabs), ~235 MB total.
  const size_t MBS = (size_t)B_ * S_;
  char* wp = (char*)d_ws;
  auto carve = [&](size_t bytes) {
    char* r = wp;
    wp += (bytes + 255) & ~(size_t)255;
    return r;
  };
  float* x      = (float*)carve(MBS * D_ * 4);
  float* tmp    = (float*)carve(MBS * D_ * 4);
  float* emis   = (float*)carve(MBS * T_ * 4);
  float* scoreb = (float*)carve(B_ * 4);
  float* logzb  = (float*)carve(B_ * 4);
  unsigned short* xbf    = (unsigned short*)carve(MBS * D_ * 2);
  unsigned short* qkvbf  = (unsigned short*)carve(MBS * 3 * D_ * 2);
  unsigned short* obf    = (unsigned short*)carve(MBS * D_ * 2);
  unsigned short* hbf    = (unsigned short*)carve(MBS * FF_ * 2);
  unsigned short* WqkvB  = (unsigned short*)carve((size_t)L_ * 3 * D_ * D_ * 2);
  unsigned short* WoB    = (unsigned short*)carve((size_t)L_ * D_ * D_ * 2);
  unsigned short* W1B    = (unsigned short*)carve((size_t)L_ * FF_ * D_ * 2);
  unsigned short* W2B    = (unsigned short*)carve((size_t)L_ * D_ * FF_ * 2);
  unsigned short* WtagB  = (unsigned short*)carve((size_t)T_ * D_ * 2);

  auto cvt = [&](const float* src, unsigned short* dst, size_t n) {
    cvt_bf16_kernel<<<(unsigned)((n / 4 + 255) / 256), 256, 0, stream>>>(src, dst, (int)n);
  };
  cvt(Wqkv, WqkvB, (size_t)L_ * 3 * D_ * D_);
  cvt(Wo,   WoB,   (size_t)L_ * D_ * D_);
  cvt(W1,   W1B,   (size_t)L_ * FF_ * D_);
  cvt(W2,   W2B,   (size_t)L_ * D_ * FF_);
  cvt(Wtag, WtagB, (size_t)T_ * D_);

  const int M = B_ * S_;  // 16384
  embed_kernel<<<M, 256, 0, stream>>>(tokens, emb, x, xbf);

  for (int l = 0; l < L_; ++l) {
    // qkv (bf16 only: consumed by attention)
    gemm_bt_kernel<<<dim3((3 * D_) / 64, M / 128), 128, 0, stream>>>(
        xbf, WqkvB + (size_t)l * 3 * D_ * D_, bqkv + (size_t)l * 3 * D_,
        nullptr, qkvbf, M, 3 * D_, D_, 0);
    attn_kernel<<<dim3(S_ / 64, B_ * H_), 128, 0, stream>>>(qkvbf, obf);
    // o @ Wo^T (f32: feeds residual+LN)
    gemm_bt_kernel<<<dim3(D_ / 64, M / 128), 128, 0, stream>>>(
        obf, WoB + (size_t)l * D_ * D_, bo + (size_t)l * D_,
        tmp, nullptr, M, D_, D_, 0);
    add_ln_kernel<<<M, 256, 0, stream>>>(x, tmp, ln1_s + (size_t)l * D_,
                                         ln1_b + (size_t)l * D_, x, xbf);
    // FFN1 + ReLU (bf16 only: consumed by FFN2)
    gemm_bt_kernel<<<dim3(FF_ / 64, M / 128), 128, 0, stream>>>(
        xbf, W1B + (size_t)l * FF_ * D_, b1 + (size_t)l * FF_,
        nullptr, hbf, M, FF_, D_, 1);
    // FFN2 (f32: feeds residual+LN)
    gemm_bt_kernel<<<dim3(D_ / 64, M / 128), 128, 0, stream>>>(
        hbf, W2B + (size_t)l * D_ * FF_, b2 + (size_t)l * D_,
        tmp, nullptr, M, D_, FF_, 0);
    add_ln_kernel<<<M, 256, 0, stream>>>(x, tmp, ln2_s + (size_t)l * D_,
                                         ln2_b + (size_t)l * D_, x, xbf);
  }

  // Emissions (N=32 < tile: TDM OOB rows read as zero; stores guarded).
  gemm_bt_kernel<<<dim3(1, M / 128), 128, 0, stream>>>(
      xbf, WtagB, btag, emis, nullptr, M, T_, D_, 0);

  crf_score_kernel<<<1, 32, 0, stream>>>(emis, tags, trans, start_t, end_t, scoreb);
  crf_fwd_kernel<<<B_, 32, 0, stream>>>(emis, trans, start_t, end_t, logzb);
  crf_loss_kernel<<<1, 32, 0, stream>>>(logzb, scoreb, (float*)d_out);
}
